// GRUCell_44659069944297
// MI455X (gfx1250) — matrix operand order repaired and checked by reading, not compiled, not run on hardware
//
#include <hip/hip_runtime.h>
#include <hip/hip_bf16.h>

// ---------------------------------------------------------------------------
// GRU on gfx1250 via v_wmma_f32_16x16x32_bf16.
// B=64, T=512, D_IN=512, D_H=1024, D_OUT=1024
// Weights pre-transposed to (N x K) bf16: A and B WMMA fragments are
// contiguous 128-bit loads. Recurrent weights LDS-resident in the scan.
// Ping-pong double-buffered inner loops (no register-rotation moves).
// ---------------------------------------------------------------------------

#define GB 64        // batch
#define GT 512       // time steps
#define GDIN 512
#define GDH 1024
#define GDOUT 1024

typedef __attribute__((ext_vector_type(16))) __bf16 v16bf;
typedef __attribute__((ext_vector_type(8)))  __bf16 v8bf;
typedef __attribute__((ext_vector_type(8)))  float  v8f;

// ---------------------------------------------------------------------------
// WMMA fragment helpers (this code's contiguous k = 16*half + s order, wave32)
// ---------------------------------------------------------------------------

// A fragment: 16x32 bf16 tile, row-major A.  p = &A[row, k0 + half*16].
__device__ __forceinline__ v16bf load_a_frag(const __bf16* p) {
    v8bf lo = *(const v8bf*)p;
    v8bf hi = *(const v8bf*)(p + 8);
    return __builtin_shufflevector(lo, hi, 0, 1, 2, 3, 4, 5, 6, 7,
                                   8, 9, 10, 11, 12, 13, 14, 15);
}

// B fragment from TRANSPOSED weights WT (N x K row-major).
// p = &WT[col, k0 + half*16] -> 16 contiguous bf16 (32B).
__device__ __forceinline__ v16bf load_b_frag_t(const __bf16* p) {
    v8bf lo = *(const v8bf*)p;
    v8bf hi = *(const v8bf*)(p + 8);
    return __builtin_shufflevector(lo, hi, 0, 1, 2, 3, 4, 5, 6, 7,
                                   8, 9, 10, 11, 12, 13, 14, 15);
}

__device__ __forceinline__ v8f wmma_bf16(v16bf a, v16bf b, v8f acc) {
    return __builtin_amdgcn_wmma_f32_16x16x32_bf16(
        /*neg_a=*/false, a, /*neg_b=*/false, b,
        /*c_mod=*/(short)0, acc, /*reuse_a=*/false, /*reuse_b=*/false);
}

__device__ __forceinline__ float sigmoidf_(float x) {
    return 1.0f / (1.0f + __expf(-x));
}

#define V8F_ZERO {0.f, 0.f, 0.f, 0.f, 0.f, 0.f, 0.f, 0.f}

// ---------------------------------------------------------------------------
// Grid-wide barrier (generation counter), agent scope.
// ---------------------------------------------------------------------------
struct GridBar { unsigned count; unsigned gen; };

__device__ __forceinline__ void grid_sync(GridBar* bar, unsigned nblk) {
    __syncthreads();
    if (threadIdx.x == 0) {
        __threadfence();
        unsigned g = __hip_atomic_load(&bar->gen, __ATOMIC_RELAXED,
                                       __HIP_MEMORY_SCOPE_AGENT);
        unsigned my = __hip_atomic_fetch_add(&bar->count, 1u, __ATOMIC_ACQ_REL,
                                             __HIP_MEMORY_SCOPE_AGENT);
        if (my == nblk - 1u) {
            __hip_atomic_store(&bar->count, 0u, __ATOMIC_RELAXED,
                               __HIP_MEMORY_SCOPE_AGENT);
            __hip_atomic_fetch_add(&bar->gen, 1u, __ATOMIC_RELEASE,
                                   __HIP_MEMORY_SCOPE_AGENT);
        } else {
            while (__hip_atomic_load(&bar->gen, __ATOMIC_ACQUIRE,
                                     __HIP_MEMORY_SCOPE_AGENT) == g) {
                __builtin_amdgcn_s_sleep(1);
            }
        }
        __threadfence();
    }
    __syncthreads();
}

__global__ void init_bar_kernel(GridBar* bar) {
    if (threadIdx.x == 0) { bar->count = 0u; bar->gen = 0u; }
}

// ---------------------------------------------------------------------------
// fp32 -> bf16 conversion (grid-stride)
// ---------------------------------------------------------------------------
__global__ void cvt_bf16_kernel(const float* __restrict__ src,
                                __bf16* __restrict__ dst, size_t n) {
    size_t i = (size_t)blockIdx.x * blockDim.x + threadIdx.x;
    size_t stride = (size_t)gridDim.x * blockDim.x;
    for (; i < n; i += stride) dst[i] = (__bf16)src[i];
}

// fp32 (K x 1024 row-major) -> bf16 transposed (1024 x K row-major).
__global__ void cvt_t1024_kernel(const float* __restrict__ src,
                                 __bf16* __restrict__ dst, int K) {
    size_t total = (size_t)K * 1024;
    size_t i = (size_t)blockIdx.x * blockDim.x + threadIdx.x;
    size_t stride = (size_t)gridDim.x * blockDim.x;
    for (; i < total; i += stride) {
        size_t k = i >> 10;
        size_t n = i & 1023;
        dst[n * (size_t)K + k] = (__bf16)src[i];
    }
}

// ---------------------------------------------------------------------------
// bf16 GEMM: C = act(A[M,K] @ B[K,N] + bias[N]); B given TRANSPOSED (NxK).
// Block = 256 threads (8 waves), 64x128 tile.
// Wave w: row tile mt = w&3, col tiles (w>>2)*4 + {0..3}.
// Ping-pong pipelined (unroll x2): operands are loaded one stage ahead.
// If tmajor != 0, logical row (b*GT + t) is stored at (t*GB + b) (time-major).
// M % 64 == 0, N % 128 == 0, K % 64 == 0. A/BT readable 64B past end.
// ---------------------------------------------------------------------------
__global__ void __launch_bounds__(256)
wmma_gemm_kernel(const __bf16* __restrict__ A, const __bf16* __restrict__ BT,
                 const float* __restrict__ bias, float* __restrict__ C,
                 int M, int N, int K, int act, int tmajor) {
    const int wave = threadIdx.x >> 5;
    const int lane = threadIdx.x & 31;
    const int r = lane & 15;
    const int half = lane >> 4;
    const int mt = wave & 3;
    const int ng = wave >> 2;           // 0 or 1

    const int row0 = blockIdx.y * 64 + mt * 16;
    const int colBase = blockIdx.x * 128 + ng * 64;   // 4 tiles of 16

    const __bf16* aPtr = A + (size_t)(row0 + r) * K + half * 16;
    const __bf16* bPtr[4];
#pragma unroll
    for (int j = 0; j < 4; ++j)
        bPtr[j] = BT + (size_t)(colBase + j * 16 + r) * K + half * 16;

    v8f acc[4] = {V8F_ZERO, V8F_ZERO, V8F_ZERO, V8F_ZERO};

    // ping-pong pipeline: buf0 holds K-step k0, buf1 holds k0+32
    v16bf a0 = load_a_frag(aPtr);
    v16bf b0[4];
#pragma unroll
    for (int j = 0; j < 4; ++j) b0[j] = load_b_frag_t(bPtr[j]);

    for (int k0 = 0; k0 < K; k0 += 64) {
        v16bf a1 = load_a_frag(aPtr + k0 + 32);           // prefetch stage 1
        v16bf b1[4];
#pragma unroll
        for (int j = 0; j < 4; ++j) b1[j] = load_b_frag_t(bPtr[j] + k0 + 32);
#pragma unroll
        for (int j = 0; j < 4; ++j) acc[j] = wmma_bf16(a0, b0[j], acc[j]);

        a0 = load_a_frag(aPtr + k0 + 64);                 // prefetch stage 0
#pragma unroll
        for (int j = 0; j < 4; ++j) b0[j] = load_b_frag_t(bPtr[j] + k0 + 64);
#pragma unroll
        for (int j = 0; j < 4; ++j) acc[j] = wmma_bf16(a1, b1[j], acc[j]);
    }

#pragma unroll
    for (int j = 0; j < 4; ++j) {
        const int col = colBase + j * 16 + r;
        const float bv = bias ? bias[col] : 0.f;
#pragma unroll
        for (int v = 0; v < 8; ++v) {
            const int row = row0 + v + half * 8;
            // time-major permute: row = b*512 + t  ->  t*64 + b
            const int orow = tmajor ? (((row & (GT - 1)) << 6) + (row >> 9)) : row;
            float x = acc[j][v] + bv;
            if (act) x = fmaxf(x, 0.f);
            C[(size_t)orow * N + col] = x;
        }
    }
}

// ---------------------------------------------------------------------------
// Persistent recurrent scan.
// 64 blocks x 256 threads (8 waves). Block g owns columns [g*16, g*16+16).
// Wave w: row tile (w&3), K-half (w>>2) of [0,512)/[512,1024).
// Recurrent weight slices (3 x 32KB, contiguous rows of the transposed
// weights) are copied into LDS once; B fragments come from ds_load_b128.
// Partial accumulators are reduced across the two K-halves through LDS,
// so the dependent-WMMA chain per phase is 16 deep instead of 32.
// xz/xr/xs are TIME-MAJOR: index ((t*GB + b)*GDH + col).
// ---------------------------------------------------------------------------
__global__ void __launch_bounds__(256)
gru_scan_kernel(const float* __restrict__ xz, const float* __restrict__ xr,
                const float* __restrict__ xs,
                const __bf16* __restrict__ WzhT, const __bf16* __restrict__ WrhT,
                const __bf16* __restrict__ WshT,
                float* __restrict__ hF, __bf16* __restrict__ hB,
                __bf16* __restrict__ rhB, __bf16* __restrict__ hsB,
                GridBar* bar) {
    const int g    = blockIdx.x;            // column slice
    const int wave = threadIdx.x >> 5;
    const int lane = threadIdx.x & 31;
    const int r    = lane & 15;
    const int half = lane >> 4;
    const int mt   = wave & 3;              // row tile
    const int kh   = wave >> 2;             // K half
    const int row0 = mt * 16;
    const int col0 = g * 16;
    const int col  = col0 + r;
    const int kBeg = kh * (GDH / 2);
    const unsigned nblk = gridDim.x;

    __shared__ __bf16 sWz[16 * GDH];        // 32KB each: this block's 16
    __shared__ __bf16 sWr[16 * GDH];        // columns of the transposed
    __shared__ __bf16 sWs[16 * GDH];        // recurrent weights (row-contig)
    __shared__ float  sAcc[2][4][256];      // [gate][rowTile][lane*8+v], 8KB

    // ---- stage weight slices into LDS (each slice is contiguous) ----
    {
        const v8bf* gz = (const v8bf*)(WzhT + (size_t)col0 * GDH);
        const v8bf* gr = (const v8bf*)(WrhT + (size_t)col0 * GDH);
        const v8bf* gs = (const v8bf*)(WshT + (size_t)col0 * GDH);
        v8bf* lz = (v8bf*)sWz;
        v8bf* lr = (v8bf*)sWr;
        v8bf* ls = (v8bf*)sWs;
        for (int i = threadIdx.x; i < (16 * GDH) / 8; i += blockDim.x) {
            lz[i] = gz[i];
            lr[i] = gr[i];
            ls[i] = gs[i];
        }
    }

    const __bf16* aBaseH  = hB  + (size_t)(row0 + r) * GDH + half * 16;
    const __bf16* aBaseRH = rhB + (size_t)(row0 + r) * GDH + half * 16;
    const __bf16* sWzp = sWz + (size_t)r * GDH + half * 16;
    const __bf16* sWrp = sWr + (size_t)r * GDH + half * 16;
    const __bf16* sWsp = sWs + (size_t)r * GDH + half * 16;

    // zero-init hidden state for this block's column slice
    for (int i = threadIdx.x; i < GB * 16; i += blockDim.x) {
        const int b = i >> 4;
        const int c = col0 + (i & 15);
        hF[(size_t)b * GDH + c] = 0.f;
        hB[(size_t)b * GDH + c] = (__bf16)0.f;
    }
    grid_sync(bar, nblk);

    float zv[8];

    for (int t = 0; t < GT; ++t) {
        // ---- phase 1: z and r partial GEMMs over this wave's K-half ----
        // hoist gate-input loads so they overlap with the WMMA loop
        float xzv[8], xrv[8];
        if (kh == 0) {
#pragma unroll
            for (int v = 0; v < 8; ++v) {
                const int b = row0 + v + half * 8;
                const size_t xi = ((size_t)t * GB + b) * GDH + col;
                xzv[v] = xz[xi];
                xrv[v] = xr[xi];
            }
        }
        v8f accZ = V8F_ZERO;
        v8f accR = V8F_ZERO;
        {
            v16bf a0 = load_a_frag(aBaseH + kBeg);        // ping-pong pipeline
            for (int k0 = kBeg; k0 < kBeg + GDH / 2; k0 += 64) {
                v16bf a1  = load_a_frag(aBaseH + k0 + 32);
                v16bf bz0 = load_b_frag_t(sWzp + k0);     // LDS
                v16bf br0 = load_b_frag_t(sWrp + k0);     // LDS
                accZ = wmma_bf16(a0, bz0, accZ);
                accR = wmma_bf16(a0, br0, accR);
                a0 = load_a_frag(aBaseH + k0 + 64);
                v16bf bz1 = load_b_frag_t(sWzp + k0 + 32);
                v16bf br1 = load_b_frag_t(sWrp + k0 + 32);
                accZ = wmma_bf16(a1, bz1, accZ);
                accR = wmma_bf16(a1, br1, accR);
            }
        }
        if (kh == 1) {
#pragma unroll
            for (int v = 0; v < 8; ++v) {
                sAcc[0][mt][lane * 8 + v] = accZ[v];
                sAcc[1][mt][lane * 8 + v] = accR[v];
            }
        }
        __syncthreads();
        if (kh == 0) {
#pragma unroll
            for (int v = 0; v < 8; ++v) {
                const int b = row0 + v + half * 8;
                const size_t hi = (size_t)b * GDH + col;
                const float z  = sigmoidf_(accZ[v] + sAcc[0][mt][lane * 8 + v] + xzv[v]);
                const float rr = sigmoidf_(accR[v] + sAcc[1][mt][lane * 8 + v] + xrv[v]);
                zv[v] = z;
                rhB[hi] = (__bf16)(rr * hF[hi]);
            }
        }
        grid_sync(bar, nblk);

        // ---- phase 2: s~ = tanh((r*h) @ Wsh + xs_t), state update ----
        float xsv[8];
        if (kh == 0) {
#pragma unroll
            for (int v = 0; v < 8; ++v) {
                const int b = row0 + v + half * 8;
                xsv[v] = xs[((size_t)t * GB + b) * GDH + col];
            }
        }
        v8f accS = V8F_ZERO;
        {
            v16bf a0 = load_a_frag(aBaseRH + kBeg);       // ping-pong pipeline
            for (int k0 = kBeg; k0 < kBeg + GDH / 2; k0 += 64) {
                v16bf a1  = load_a_frag(aBaseRH + k0 + 32);
                v16bf bs0 = load_b_frag_t(sWsp + k0);     // LDS
                accS = wmma_bf16(a0, bs0, accS);
                a0 = load_a_frag(aBaseRH + k0 + 64);
                v16bf bs1 = load_b_frag_t(sWsp + k0 + 32);
                accS = wmma_bf16(a1, bs1, accS);
            }
        }
        if (kh == 1) {
#pragma unroll
            for (int v = 0; v < 8; ++v)
                sAcc[0][mt][lane * 8 + v] = accS[v];
        }
        __syncthreads();
        if (kh == 0) {
#pragma unroll
            for (int v = 0; v < 8; ++v) {
                const int b = row0 + v + half * 8;
                const size_t hi = (size_t)b * GDH + col;
                const float sv   = tanhf(accS[v] + sAcc[0][mt][lane * 8 + v] + xsv[v]);
                const float hold = hF[hi];
                const float hn   = (1.f - zv[v]) * hold + zv[v] * sv;
                hF[hi]  = hn;
                hB[hi]  = (__bf16)hn;
                hsB[((size_t)b * GT + t) * GDH + col] = (__bf16)hn;
            }
        }
        grid_sync(bar, nblk);
    }
}

// ---------------------------------------------------------------------------
// copy h_final (fp32) to the tail of d_out
// ---------------------------------------------------------------------------
__global__ void copy_hfinal_kernel(const float* __restrict__ hF,
                                   float* __restrict__ out) {
    int i = blockIdx.x * blockDim.x + threadIdx.x;
    if (i < GB * GDH) out[i] = hF[i];
}

// ---------------------------------------------------------------------------
// Launch
// ---------------------------------------------------------------------------
extern "C" void kernel_launch(void* const* d_in, const int* in_sizes, int n_in,
                              void* d_out, int out_size, void* d_ws, size_t ws_size,
                              hipStream_t stream) {
    (void)in_sizes; (void)n_in; (void)out_size; (void)ws_size;

    const float* inputs = (const float*)d_in[0];
    const float* Wz = (const float*)d_in[1];
    const float* bz = (const float*)d_in[2];
    const float* Wr = (const float*)d_in[3];
    const float* br = (const float*)d_in[4];
    const float* Ws = (const float*)d_in[5];
    const float* bs = (const float*)d_in[6];
    const float* Wo = (const float*)d_in[7];
    const float* bo = (const float*)d_in[8];
    float* out = (float*)d_out;

    // workspace carve-out (all regions 256B-aligned; +64B pads cover the
    // software-pipeline prefetch overrun, whose value is never consumed)
    char* ws = (char*)d_ws;
    size_t off = 0;
    auto alloc = [&](size_t bytes) -> void* {
        void* p = ws + off;
        off += (bytes + 255) & ~(size_t)255;
        return p;
    };
    const size_t nIn = (size_t)GB * GT * GDIN;    // 16.8M
    const size_t nBT = (size_t)GB * GT * GDH;     // 33.6M

    __bf16* inB   = (__bf16*)alloc(nIn * 2 + 64);
    __bf16* WzxT  = (__bf16*)alloc((size_t)GDIN * GDH * 2 + 64);  // 1024x512
    __bf16* WrxT  = (__bf16*)alloc((size_t)GDIN * GDH * 2 + 64);
    __bf16* WsxT  = (__bf16*)alloc((size_t)GDIN * GDH * 2 + 64);
    __bf16* WzhT  = (__bf16*)alloc((size_t)GDH * GDH * 2 + 64);   // 1024x1024
    __bf16* WrhT  = (__bf16*)alloc((size_t)GDH * GDH * 2 + 64);
    __bf16* WshT  = (__bf16*)alloc((size_t)GDH * GDH * 2 + 64);
    __bf16* WoT   = (__bf16*)alloc((size_t)GDH * GDOUT * 2 + 64); // 1024x1024
    float*  xzF   = (float*)alloc(nBT * 4);       // TIME-MAJOR (t, B, DH)
    float*  xrF   = (float*)alloc(nBT * 4);
    float*  xsF   = (float*)alloc(nBT * 4);
    __bf16* hsB   = (__bf16*)alloc(nBT * 2 + 64); // (b, t, h)
    float*  hF    = (float*)alloc((size_t)GB * GDH * 4);
    __bf16* hB    = (__bf16*)alloc((size_t)GB * GDH * 2 + 64);
    __bf16* rhB   = (__bf16*)alloc((size_t)GB * GDH * 2 + 64);
    GridBar* bar  = (GridBar*)alloc(sizeof(GridBar));

    // --- conversions: inputs plain, weights transposed to (N x K) bf16 ---
    cvt_bf16_kernel<<<2048, 256, 0, stream>>>(inputs, inB, nIn);
    // W[:DH] = hidden rows, W[DH:] = input rows; row-major (DIN+DH, DH=1024)
    cvt_t1024_kernel<<<512, 256, 0, stream>>>(Wz, WzhT, GDH);
    cvt_t1024_kernel<<<512, 256, 0, stream>>>(Wr, WrhT, GDH);
    cvt_t1024_kernel<<<512, 256, 0, stream>>>(Ws, WshT, GDH);
    cvt_t1024_kernel<<<256, 256, 0, stream>>>(Wz + (size_t)GDH * GDH, WzxT, GDIN);
    cvt_t1024_kernel<<<256, 256, 0, stream>>>(Wr + (size_t)GDH * GDH, WrxT, GDIN);
    cvt_t1024_kernel<<<256, 256, 0, stream>>>(Ws + (size_t)GDH * GDH, WsxT, GDIN);
    cvt_t1024_kernel<<<512, 256, 0, stream>>>(Wo, WoT, GDH);

    init_bar_kernel<<<1, 32, 0, stream>>>(bar);

    // --- input projections (stored time-major for the scan) ---
    {
        dim3 grid(GDH / 128, (GB * GT) / 64);
        wmma_gemm_kernel<<<grid, 256, 0, stream>>>(inB, WzxT, bz, xzF,
                                                   GB * GT, GDH, GDIN, 0, 1);
        wmma_gemm_kernel<<<grid, 256, 0, stream>>>(inB, WrxT, br, xrF,
                                                   GB * GT, GDH, GDIN, 0, 1);
        wmma_gemm_kernel<<<grid, 256, 0, stream>>>(inB, WsxT, bs, xsF,
                                                   GB * GT, GDH, GDIN, 0, 1);
    }

    // --- recurrent scan: 64 persistent blocks, grid-barrier per phase ---
    gru_scan_kernel<<<64, 256, 0, stream>>>(xzF, xrF, xsF, WzhT, WrhT, WshT,
                                            hF, hB, rhB, hsB, bar);

    // --- output projection + ReLU: (32768 x 1024) @ (1024 x 1024) ---
    {
        dim3 grid(GDOUT / 128, (GB * GT) / 64);
        wmma_gemm_kernel<<<grid, 256, 0, stream>>>(hsB, WoT, bo, out,
                                                   GB * GT, GDOUT, GDH, 1, 0);
    }

    // --- h_final tail ---
    copy_hfinal_kernel<<<(GB * GDH + 255) / 256, 256, 0, stream>>>(
        hF, out + (size_t)GB * GT * GDOUT);
}
